// PretrainGINE_37486474559541
// MI455X (gfx1250) — matrix operand order repaired and checked by reading, not compiled, not run on hardware
//
#include <hip/hip_runtime.h>
#include <hip/hip_bf16.h>

#define N_NODES   150000
#define N_EDGES   450000
#define NUM_GRAPHS 5000
#define H         256
#define L         5
#define JK_DIM    1536

typedef __bf16 bf16_t;
typedef __attribute__((ext_vector_type(16))) __bf16 v16bf;
typedef __attribute__((ext_vector_type(8)))  __bf16 v8bf;
typedef __attribute__((ext_vector_type(8)))  float  v8f;

static __device__ __forceinline__ bf16_t f2bf(float f) { return (bf16_t)f; }

// ---------------------------------------------------------------------------
// h[n] = atom_emb[x[n]]; JK slot 0 (bf16). One float4 per thread.
// ---------------------------------------------------------------------------
__global__ __launch_bounds__(256)
void k_embed(const int* __restrict__ x, const float* __restrict__ atom_emb,
             float* __restrict__ h, bf16_t* __restrict__ hkb)
{
    int g = blockIdx.x * blockDim.x + threadIdx.x;      // N_NODES*64 groups
    int row = g >> 6;
    int c4  = (g & 63) * 4;
    if (row >= N_NODES) return;
    float4 v = *(const float4*)(atom_emb + (size_t)x[row] * H + c4);
    *(float4*)(h + (size_t)row * H + c4) = v;
    bf16_t* kp = hkb + (size_t)row * JK_DIM + c4;
    kp[0] = f2bf(v.x); kp[1] = f2bf(v.y); kp[2] = f2bf(v.z); kp[3] = f2bf(v.w);
}

__global__ __launch_bounds__(256)
void k_init_vn(const float* __restrict__ vn0, float* __restrict__ vn)
{
    int idx = blockIdx.x * blockDim.x + threadIdx.x;
    if (idx >= NUM_GRAPHS * H) return;
    vn[idx] = vn0[idx & (H - 1)];
}

// ---------------------------------------------------------------------------
// h += vn[batch];  agg = (1+eps[i]) * h   (fused init of the segment sum)
// ---------------------------------------------------------------------------
__global__ __launch_bounds__(256)
void k_vn_add_init(float* __restrict__ h, const float* __restrict__ vn,
                   const int* __restrict__ batch, const float* __restrict__ eps,
                   int li, float* __restrict__ agg)
{
    int g = blockIdx.x * blockDim.x + threadIdx.x;
    int row = g >> 6;
    int c4  = (g & 63) * 4;
    if (row >= N_NODES) return;
    float ep = 1.0f + eps[li];
    int b = batch[row];
    float4 hv = *(float4*)(h + (size_t)row * H + c4);
    float4 vv = *(const float4*)(vn + (size_t)b * H + c4);
    hv.x += vv.x; hv.y += vv.y; hv.z += vv.z; hv.w += vv.w;
    *(float4*)(h + (size_t)row * H + c4) = hv;
    float4 av = make_float4(ep * hv.x, ep * hv.y, ep * hv.z, ep * hv.w);
    *(float4*)(agg + (size_t)row * H + c4) = av;
}

// ---------------------------------------------------------------------------
// One wave per edge: msg = relu(h[src] + bond_emb[ea]); atomic add into agg[dst]
// ---------------------------------------------------------------------------
__global__ __launch_bounds__(256)
void k_edge_scatter(const int* __restrict__ src, const int* __restrict__ dst,
                    const int* __restrict__ ea, const float* __restrict__ bond_emb,
                    const float* __restrict__ h, float* __restrict__ agg)
{
    int wid  = (blockIdx.x * blockDim.x + threadIdx.x) >> 5;
    int lane = threadIdx.x & 31;
    if (wid >= N_EDGES) return;
    int s = src[wid], d = dst[wid], a = ea[wid];
    const float4* hp = (const float4*)(h + (size_t)s * H) + lane * 2;
    const float4* ep = (const float4*)(bond_emb + (size_t)a * H) + lane * 2;
    float* ap = agg + (size_t)d * H + lane * 8;
    #pragma unroll
    for (int j = 0; j < 2; ++j) {
        float4 hv = hp[j];
        float4 ev = ep[j];
        atomicAdd(ap + j * 4 + 0, fmaxf(hv.x + ev.x, 0.0f));
        atomicAdd(ap + j * 4 + 1, fmaxf(hv.y + ev.y, 0.0f));
        atomicAdd(ap + j * 4 + 2, fmaxf(hv.z + ev.z, 0.0f));
        atomicAdd(ap + j * 4 + 3, fmaxf(hv.w + ev.w, 0.0f));
    }
}

// ---------------------------------------------------------------------------
// f32 -> bf16 elementwise (float4 per thread)
// ---------------------------------------------------------------------------
__global__ __launch_bounds__(256)
void k_to_bf16(const float* __restrict__ inp, bf16_t* __restrict__ outp, int n4)
{
    int idx = blockIdx.x * blockDim.x + threadIdx.x;
    if (idx >= n4) return;
    float4 v = ((const float4*)inp)[idx];
    bf16_t* o = outp + (size_t)idx * 4;
    o[0] = f2bf(v.x); o[1] = f2bf(v.y); o[2] = f2bf(v.z); o[3] = f2bf(v.w);
}

__global__ __launch_bounds__(256)
void k_zero(float* __restrict__ p, int n)
{
    int idx = blockIdx.x * blockDim.x + threadIdx.x;
    if (idx < n) p[idx] = 0.0f;
}

// ---------------------------------------------------------------------------
// Weight prep: Wt[n][k] = bf16(W[k][n]) for n < Nreal, else 0. Wt is [Npad][K].
// ---------------------------------------------------------------------------
__global__ __launch_bounds__(256)
void k_prep_w(const float* __restrict__ W, bf16_t* __restrict__ Wt,
              int K, int Nreal, int Npad)
{
    int idx = blockIdx.x * blockDim.x + threadIdx.x;
    if (idx >= Npad * K) return;
    int n = idx / K, k = idx - n * K;
    Wt[(size_t)n * K + k] = (n < Nreal) ? f2bf(W[(size_t)k * Nreal + n]) : f2bf(0.0f);
}

// ---------------------------------------------------------------------------
// bf16 WMMA GEMM: C[M,256] = A[M,K](bf16) * Wt[256,K](bf16, row=N col=K) + bias
// Block = 8 waves, tile 32(M) x 256(N); wave = 16x64 via 4 accum frags.
// MODE 0: f32 out. MODE 1: relu -> bf16 out.
// ---------------------------------------------------------------------------
template <int MODE>
__global__ __launch_bounds__(256)
void k_gemm(const bf16_t* __restrict__ A, const bf16_t* __restrict__ Wt,
            const float* __restrict__ bias, void* __restrict__ Cout,
            int M, int K)
{
    const int tid  = threadIdx.x;
    const int wave = tid >> 5;
    const int lane = tid & 31;
    const int half = lane >> 4;
    const int l16  = lane & 15;
    const int wm   = wave >> 2;            // 0..1
    const int wn   = wave & 3;             // 0..3
    const int row0 = blockIdx.x * 32 + wm * 16;
    const int n0   = wn * 64;

    int arow = row0 + l16;
    if (arow >= M) arow = M - 1;
    const bf16_t* aptr = A + (size_t)arow * K;

    v8f acc[4];
    #pragma unroll
    for (int j = 0; j < 4; ++j)
        #pragma unroll
        for (int r = 0; r < 8; ++r) acc[j][r] = 0.0f;

    for (int k0 = 0; k0 < K; k0 += 32) {
        // A fragment: lanes 0-15 hold K {0..7,16..23}; lanes 16-31 {8..15,24..31}
        v8bf alo = *(const v8bf*)(aptr + k0 + half * 8);
        v8bf ahi = *(const v8bf*)(aptr + k0 + 16 + half * 8);
        v16bf a;
        #pragma unroll
        for (int i = 0; i < 8; ++i) { a[i] = alo[i]; a[8 + i] = ahi[i]; }
        #pragma unroll
        for (int j = 0; j < 4; ++j) {
            int n = n0 + j * 16 + l16;
            const bf16_t* bptr = Wt + (size_t)n * K + k0 + half * 8;
            v8bf blo = *(const v8bf*)(bptr);
            v8bf bhi = *(const v8bf*)(bptr + 16);
            v16bf b;
            #pragma unroll
            for (int i = 0; i < 8; ++i) { b[i] = blo[i]; b[8 + i] = bhi[i]; }
            acc[j] = __builtin_amdgcn_wmma_f32_16x16x32_bf16(
                false, a, false, b, (short)0, acc[j], false, false);
        }
    }
    #pragma unroll
    for (int j = 0; j < 4; ++j) {
        int col = n0 + j * 16 + l16;
        float bv = bias[col];
        #pragma unroll
        for (int r = 0; r < 8; ++r) {
            int row = row0 + r + half * 8;
            if (row < M) {
                float v = acc[j][r] + bv;
                if (MODE == 1) {
                    v = fmaxf(v, 0.0f);
                    ((bf16_t*)Cout)[(size_t)row * H + col] = f2bf(v);
                } else {
                    ((float*)Cout)[(size_t)row * H + col] = v;
                }
            }
        }
    }
}

// ---------------------------------------------------------------------------
// Node epilogue: h = relu(LN(z2))*... + h_prev; write h(f32), JK slot (bf16),
// and atomically accumulate vn_agg[batch]. One wave per row.
// ---------------------------------------------------------------------------
__global__ __launch_bounds__(256)
void k_ln_node(const float* __restrict__ z2, const float* __restrict__ g,
               const float* __restrict__ bsh, float* __restrict__ h,
               bf16_t* __restrict__ hkb, int slot, const int* __restrict__ batch,
               float* __restrict__ vn_agg)
{
    int row  = (blockIdx.x * blockDim.x + threadIdx.x) >> 5;
    int lane = threadIdx.x & 31;
    if (row >= N_NODES) return;
    const float* zp = z2 + (size_t)row * H + lane * 8;
    float v[8], s = 0.f, ss = 0.f;
    #pragma unroll
    for (int i = 0; i < 8; ++i) { v[i] = zp[i]; s += v[i]; ss += v[i] * v[i]; }
    #pragma unroll
    for (int off = 16; off; off >>= 1) { s += __shfl_xor(s, off, 32); ss += __shfl_xor(ss, off, 32); }
    float mean = s * (1.0f / H);
    float var  = ss * (1.0f / H) - mean * mean;
    float rstd = rsqrtf(var + 1e-5f);
    float* hp = h + (size_t)row * H + lane * 8;
    bf16_t* kp = hkb + (size_t)row * JK_DIM + (size_t)slot * H + lane * 8;
    float* vp = vn_agg + (size_t)batch[row] * H + lane * 8;
    const float* gp = g + lane * 8;
    const float* bp = bsh + lane * 8;
    #pragma unroll
    for (int i = 0; i < 8; ++i) {
        float y  = fmaxf((v[i] - mean) * rstd * gp[i] + bp[i], 0.0f);
        float hn = y + hp[i];
        hp[i] = hn;
        kp[i] = f2bf(hn);
        atomicAdd(vp + i, hn);
    }
}

// sb = bf16(vn_agg + vn)
__global__ __launch_bounds__(256)
void k_vn_prep(const float* __restrict__ vn_agg, const float* __restrict__ vn,
               bf16_t* __restrict__ sb)
{
    int idx = blockIdx.x * blockDim.x + threadIdx.x;
    if (idx >= NUM_GRAPHS * H) return;
    sb[idx] = f2bf(vn_agg[idx] + vn[idx]);
}

// vn = relu(LN(vtmp)); one wave per row
__global__ __launch_bounds__(256)
void k_ln_vn(const float* __restrict__ z, const float* __restrict__ g,
             const float* __restrict__ b, float* __restrict__ vnout)
{
    int row  = (blockIdx.x * blockDim.x + threadIdx.x) >> 5;
    int lane = threadIdx.x & 31;
    if (row >= NUM_GRAPHS) return;
    const float* zp = z + (size_t)row * H + lane * 8;
    float v[8], s = 0.f, ss = 0.f;
    #pragma unroll
    for (int i = 0; i < 8; ++i) { v[i] = zp[i]; s += v[i]; ss += v[i] * v[i]; }
    #pragma unroll
    for (int off = 16; off; off >>= 1) { s += __shfl_xor(s, off, 32); ss += __shfl_xor(ss, off, 32); }
    float mean = s * (1.0f / H);
    float var  = ss * (1.0f / H) - mean * mean;
    float rstd = rsqrtf(var + 1e-5f);
    float* op = vnout + (size_t)row * H + lane * 8;
    const float* gp = g + lane * 8;
    const float* bp = b + lane * 8;
    #pragma unroll
    for (int i = 0; i < 8; ++i)
        op[i] = fmaxf((v[i] - mean) * rstd * gp[i] + bp[i], 0.0f);
}

// ---------------------------------------------------------------------------
// Final projection: out[M,28] = th[M,256](bf16) @ W2 + b2 via WMMA, N padded 32.
// Block = 2 waves; wave w handles 16x16 tile at n0 = w*16; masked store n<28.
// ---------------------------------------------------------------------------
__global__ __launch_bounds__(64)
void k_headproj(const bf16_t* __restrict__ A, const bf16_t* __restrict__ Wt,
                const float* __restrict__ bias, float* __restrict__ out, int M)
{
    const int tid  = threadIdx.x;
    const int wave = tid >> 5;             // 0..1
    const int lane = tid & 31;
    const int half = lane >> 4;
    const int l16  = lane & 15;
    const int row0 = blockIdx.x * 16;
    const int n0   = wave * 16;
    int arow = row0 + l16;
    if (arow >= M) arow = M - 1;
    const bf16_t* aptr = A + (size_t)arow * H;
    v8f acc;
    #pragma unroll
    for (int r = 0; r < 8; ++r) acc[r] = 0.0f;
    for (int k0 = 0; k0 < H; k0 += 32) {
        v8bf alo = *(const v8bf*)(aptr + k0 + half * 8);
        v8bf ahi = *(const v8bf*)(aptr + k0 + 16 + half * 8);
        v16bf a;
        #pragma unroll
        for (int i = 0; i < 8; ++i) { a[i] = alo[i]; a[8 + i] = ahi[i]; }
        const bf16_t* bptr = Wt + (size_t)(n0 + l16) * H + k0 + half * 8;
        v8bf blo = *(const v8bf*)(bptr);
        v8bf bhi = *(const v8bf*)(bptr + 16);
        v16bf b;
        #pragma unroll
        for (int i = 0; i < 8; ++i) { b[i] = blo[i]; b[8 + i] = bhi[i]; }
        acc = __builtin_amdgcn_wmma_f32_16x16x32_bf16(
            false, a, false, b, (short)0, acc, false, false);
    }
    int col = n0 + l16;
    if (col < 28) {
        float bv = bias[col];
        #pragma unroll
        for (int r = 0; r < 8; ++r) {
            int row = row0 + r + half * 8;
            if (row < M) out[(size_t)row * 28 + col] = acc[r] + bv;
        }
    }
}

// ---------------------------------------------------------------------------
extern "C" void kernel_launch(void* const* d_in, const int* in_sizes, int n_in,
                              void* d_out, int out_size, void* d_ws, size_t ws_size,
                              hipStream_t stream)
{
    (void)in_sizes; (void)n_in; (void)out_size; (void)ws_size;
    const int*   x          = (const int*)d_in[0];
    const int*   edge_index = (const int*)d_in[1];
    const int*   edge_attr  = (const int*)d_in[2];
    const int*   batch      = (const int*)d_in[3];
    const float* atom_emb   = (const float*)d_in[4];
    const float* bond_emb   = (const float*)d_in[5];
    const float* vn_emb0    = (const float*)d_in[6];
    const float* mlp_w1     = (const float*)d_in[7];
    const float* mlp_b1     = (const float*)d_in[8];
    const float* mlp_w2     = (const float*)d_in[9];
    const float* mlp_b2     = (const float*)d_in[10];
    const float* eps        = (const float*)d_in[11];
    const float* ln_g       = (const float*)d_in[12];
    const float* ln_b       = (const float*)d_in[13];
    const float* vn_w       = (const float*)d_in[14];
    const float* vn_b       = (const float*)d_in[15];
    const float* vn_ln_g    = (const float*)d_in[16];
    const float* vn_ln_b    = (const float*)d_in[17];
    const float* head_w1    = (const float*)d_in[18];
    const float* head_b1    = (const float*)d_in[19];
    const float* head_w2    = (const float*)d_in[20];
    const float* head_b2    = (const float*)d_in[21];

    const int* srcI = edge_index;
    const int* dstI = edge_index + N_EDGES;

    char* ws = (char*)d_ws;
    size_t off = 0;
    auto take = [&](size_t bytes) {
        char* p = ws + off;
        off += (bytes + 255) & ~(size_t)255;
        return p;
    };
    float*  h      = (float*) take((size_t)N_NODES * H * 4);
    float*  agg    = (float*) take((size_t)N_NODES * H * 4);   // aliases z2
    bf16_t* zb     = (bf16_t*)take((size_t)N_NODES * H * 2);   // aliases th
    bf16_t* t      = (bf16_t*)take((size_t)N_NODES * H * 2);
    bf16_t* hkb    = (bf16_t*)take((size_t)N_NODES * JK_DIM * 2);
    float*  vn     = (float*) take((size_t)NUM_GRAPHS * H * 4);
    float*  vn_agg = (float*) take((size_t)NUM_GRAPHS * H * 4);
    bf16_t* sb     = (bf16_t*)take((size_t)NUM_GRAPHS * H * 2);
    float*  vtmp   = (float*) take((size_t)NUM_GRAPHS * H * 4);
    bf16_t* w1t    = (bf16_t*)take((size_t)H * H * 2);
    bf16_t* w2t    = (bf16_t*)take((size_t)H * H * 2);
    bf16_t* vnwt   = (bf16_t*)take((size_t)H * H * 2);
    bf16_t* hw1t   = (bf16_t*)take((size_t)H * JK_DIM * 2);
    bf16_t* hw2t   = (bf16_t*)take((size_t)32 * H * 2);
    float*  z2 = agg;
    bf16_t* th = zb;

    const int TPB = 256;
    int gEmb  = (N_NODES * 64 + TPB - 1) / TPB;
    int gEdge = (N_EDGES * 32 + TPB - 1) / TPB;
    int gCvt  = (N_NODES * H / 4 + TPB - 1) / TPB;
    int gLN   = (N_NODES * 32 + TPB - 1) / TPB;
    int gVnE  = (NUM_GRAPHS * H + TPB - 1) / TPB;
    int gVnLN = (NUM_GRAPHS * 32 + TPB - 1) / TPB;
    int gGemmN = (N_NODES + 31) / 32;
    int gGemmV = (NUM_GRAPHS + 31) / 32;
    int gPrepH = (H * H + TPB - 1) / TPB;

    k_embed<<<gEmb, TPB, 0, stream>>>(x, atom_emb, h, hkb);
    k_init_vn<<<gVnE, TPB, 0, stream>>>(vn_emb0, vn);

    for (int i = 0; i < L; ++i) {
        k_prep_w<<<gPrepH, TPB, 0, stream>>>(mlp_w1 + (size_t)i * H * H, w1t, H, H, H);
        k_prep_w<<<gPrepH, TPB, 0, stream>>>(mlp_w2 + (size_t)i * H * H, w2t, H, H, H);
        k_prep_w<<<gPrepH, TPB, 0, stream>>>(vn_w  + (size_t)i * H * H, vnwt, H, H, H);

        k_vn_add_init<<<gEmb, TPB, 0, stream>>>(h, vn, batch, eps, i, agg);
        k_edge_scatter<<<gEdge, TPB, 0, stream>>>(srcI, dstI, edge_attr, bond_emb, h, agg);
        k_to_bf16<<<gCvt, TPB, 0, stream>>>(agg, zb, N_NODES * H / 4);

        k_gemm<1><<<gGemmN, TPB, 0, stream>>>(zb, w1t, mlp_b1 + (size_t)i * H, (void*)t, N_NODES, H);
        k_gemm<0><<<gGemmN, TPB, 0, stream>>>(t, w2t, mlp_b2 + (size_t)i * H, (void*)z2, N_NODES, H);

        k_zero<<<gVnE, TPB, 0, stream>>>(vn_agg, NUM_GRAPHS * H);
        k_ln_node<<<gLN, TPB, 0, stream>>>(z2, ln_g + (size_t)i * H, ln_b + (size_t)i * H,
                                           h, hkb, i + 1, batch, vn_agg);

        k_vn_prep<<<gVnE, TPB, 0, stream>>>(vn_agg, vn, sb);
        k_gemm<0><<<gGemmV, TPB, 0, stream>>>(sb, vnwt, vn_b + (size_t)i * H, (void*)vtmp, NUM_GRAPHS, H);
        k_ln_vn<<<gVnLN, TPB, 0, stream>>>(vtmp, vn_ln_g + (size_t)i * H, vn_ln_b + (size_t)i * H, vn);
    }

    // Head
    k_prep_w<<<(H * JK_DIM + TPB - 1) / TPB, TPB, 0, stream>>>(head_w1, hw1t, JK_DIM, H, H);
    k_prep_w<<<(32 * H + TPB - 1) / TPB, TPB, 0, stream>>>(head_w2, hw2t, H, 28, 32);

    k_gemm<1><<<gGemmN, TPB, 0, stream>>>(hkb, hw1t, head_b1, (void*)th, N_NODES, JK_DIM);
    k_headproj<<<(N_NODES + 15) / 16, 64, 0, stream>>>(th, hw2t, head_b2, (float*)d_out, N_NODES);
}
